// TreeMinLoss_25211458028170
// MI455X (gfx1250) — matrix operand order repaired and checked by reading, not compiled, not run on hardware
//
#include <hip/hip_runtime.h>

typedef __attribute__((ext_vector_type(16))) _Float16 v16h;
typedef __attribute__((ext_vector_type(8)))  _Float16 v8h;
typedef __attribute__((ext_vector_type(8)))  float    v8f;

#define BB 64
#define CC 1024

// ---------------------------------------------------------------- sigmoid + f16 copy
__global__ void k_sigmoid(const float* __restrict__ h, float* __restrict__ x,
                          _Float16* __restrict__ xh, int n) {
  int i = blockIdx.x * blockDim.x + threadIdx.x;
  if (i < n) {
    float v = 1.0f / (1.0f + expf(-h[i]));
    x[i]  = v;
    xh[i] = (_Float16)v;
  }
}

// -------------------------------------------- R -> Rt (f16 transpose) + row bitmasks
// maskR[i*32 + w] bit j : R[i][32w+j] != 0
__global__ void k_convR(const float* __restrict__ R, _Float16* __restrict__ Rt,
                        unsigned* __restrict__ maskR) {
  int t = blockIdx.x * 256 + threadIdx.x;   // t = i*1024 + j, coalesced read
  int i = t >> 10, j = t & (CC - 1);
  float r = R[t];
  Rt[j * CC + i] = (_Float16)r;             // scattered transpose write (tiny, L2-resident)
  unsigned m = __builtin_amdgcn_ballot_w32(r != 0.0f);
  if ((threadIdx.x & 31) == 0) maskR[i * 32 + (j >> 5)] = m;
}

// -------------------------------- maskRT from Rt (coalesced): bit j of row i = R[j][i]
__global__ void k_maskT(const _Float16* __restrict__ Rt, unsigned* __restrict__ maskRT) {
  int t = blockIdx.x * 256 + threadIdx.x;   // t = i*1024 + j over Rt
  int i = t >> 10, j = t & (CC - 1);
  float r = (float)Rt[t];
  unsigned m = __builtin_amdgcn_ballot_w32(r != 0.0f);
  if ((threadIdx.x & 31) == 0) maskRT[i * 32 + (j >> 5)] = m;
}

// ------------------------------------------------- paths = xh @ R via WMMA f16->f32
// grid = 64 blocks (N tiles), 128 threads = 4 waves (M tiles). Full EXEC everywhere.
__global__ void __launch_bounds__(128, 1)
k_paths_wmma(const _Float16* __restrict__ xh, const _Float16* __restrict__ Rt,
             float* __restrict__ paths) {
  int lane = threadIdx.x & 31;
  int wave = threadIdx.x >> 5;
  int lh = lane & 15;         // row-within-tile selector
  int hh = lane >> 4;         // half-wave selector
  int m0 = wave * 16;
  int n0 = blockIdx.x * 16;

  // A: row m0+lh of xh; per lane: halves [base..base+7] and [base+8..base+15], base=hh*16
  const _Float16* arow = xh + (m0 + lh) * CC + hh * 16;
  // B: Rt is R^T (N-major); per lane: column n0+lh, 16 consecutive K starting at hh*16
  const _Float16* brow = Rt + (n0 + lh) * CC + hh * 16;

  v8f acc = {};
  for (int k0 = 0; k0 < CC; k0 += 32) {
    v8h alo = *(const v8h*)(arow + k0);
    v8h ahi = *(const v8h*)(arow + k0 + 8);
    v16h a = __builtin_shufflevector(alo, ahi, 0, 1, 2, 3, 4, 5, 6, 7,
                                               8, 9, 10, 11, 12, 13, 14, 15);
    v16h b = *(const v16h*)(brow + k0);      // 32B aligned contiguous load
    __builtin_prefetch((const void*)(brow + k0 + 128), 0, 0);  // global_prefetch_b8
    acc = __builtin_amdgcn_wmma_f32_16x16x32_f16(
        /*neg_a=*/false, a, /*neg_b=*/false, b,
        /*c_mod=*/(short)0, acc, /*reuse_a=*/false, /*reuse_b=*/false);
  }

  // C/D layout: lane -> N = lh ; vgpr r -> M = hh*8 + r
  float* crow = paths + (m0 + hh * 8) * CC + n0 + lh;
#pragma unroll
  for (int r = 0; r < 8; ++r) crow[r * CC] = acc[r];
}

// ---------------------------------------------------------------- per-row argmax
__global__ void k_argmax(const float* __restrict__ paths, int* __restrict__ idx) {
  __shared__ float sv[256];
  __shared__ int   si[256];
  int b = blockIdx.x, t = threadIdx.x;
  const float* row = paths + b * CC;
  float bv = -__builtin_inff();
  int bi = 0;
  for (int j = t; j < CC; j += 256) {
    float v = row[j];
    if (v > bv) { bv = v; bi = j; }          // strict > keeps first max per thread
  }
  sv[t] = bv; si[t] = bi;
  __syncthreads();
  for (int s = 128; s > 0; s >>= 1) {
    if (t < s) {
      float ov = sv[t + s]; int oi = si[t + s];
      if (ov > sv[t] || (ov == sv[t] && oi < si[t])) { sv[t] = ov; si[t] = oi; }
    }
    __syncthreads();
  }
  if (t == 0) idx[b] = si[0];
}

// ------------------- mcm / pv via bitmask scans + probs + BCE partial per block
// grid = (4, 64): blockIdx.y = b, block covers 256 classes. x row staged in LDS.
__global__ void __launch_bounds__(256)
k_loss(const float* __restrict__ x, const float* __restrict__ y,
       const unsigned* __restrict__ maskR, const unsigned* __restrict__ maskRT,
       const int* __restrict__ idx, float* __restrict__ out,
       float* __restrict__ partial) {
  __shared__ float xs[CC];
  __shared__ float red[256];
  int b = blockIdx.y;
  int i = blockIdx.x * 256 + threadIdx.x;
  const float* xr = x + b * CC;
  for (int j = threadIdx.x; j < CC; j += 256) xs[j] = xr[j];
  __syncthreads();

  float mx = 0.0f;      // mcm: max over ancestor set (diag guarantees non-empty)
  float mn = 1e30f;     // pv : min over descendant set (sentinel semantics, x > 0)
  const unsigned* wm = maskR  + i * 32;
  const unsigned* wt = maskRT + i * 32;
  for (int w = 0; w < 32; ++w) {
    unsigned m = wm[w];
    while (m) { int j = __builtin_ctz(m); m &= m - 1; mx = fmaxf(mx, xs[w * 32 + j]); }
    unsigned tm = wt[w];
    while (tm) { int j = __builtin_ctz(tm); tm &= tm - 1; mn = fminf(mn, xs[w * 32 + j]); }
  }

  int ib = idx[b];
  // probs[b][i] = R[i][idx_b] * x[b][i]
  float pr = ((wm[ib >> 5] >> (ib & 31)) & 1u) ? xs[i] : 0.0f;
  out[1 + b * CC + i] = pr;

  float yv = y[b * CC + i];
  float hstar = (1.0f - yv) * mx + mn * yv;
  float lp  = fmaxf(logf(hstar), -100.0f);
  float l1p = fmaxf(log1pf(-hstar), -100.0f);
  float term = yv * lp + (1.0f - yv) * l1p;

  red[threadIdx.x] = term;
  __syncthreads();
  for (int s = 128; s > 0; s >>= 1) {
    if (threadIdx.x < s) red[threadIdx.x] += red[threadIdx.x + s];
    __syncthreads();
  }
  if (threadIdx.x == 0) partial[blockIdx.y * gridDim.x + blockIdx.x] = red[0];
}

// ---------------------------------------------------------------- final reduce
__global__ void k_final(const float* __restrict__ partial, float* __restrict__ out) {
  __shared__ float red[256];
  int t = threadIdx.x;
  red[t] = partial[t];
  __syncthreads();
  for (int s = 128; s > 0; s >>= 1) {
    if (t < s) red[t] += red[t + s];
    __syncthreads();
  }
  if (t == 0) out[0] = -red[0] * (1.0f / (float)(BB * CC));
}

extern "C" void kernel_launch(void* const* d_in, const int* in_sizes, int n_in,
                              void* d_out, int out_size, void* d_ws, size_t ws_size,
                              hipStream_t stream) {
  const float* h = (const float*)d_in[0];   // (64,1024)
  const float* y = (const float*)d_in[1];   // (64,1024)
  const float* R = (const float*)d_in[2];   // (1024,1024)

  char* ws = (char*)d_ws;                   // ~3.0 MB total, all 256B-aligned offsets
  float*    x      = (float*)   (ws + 0);        // 256 KB
  _Float16* xh     = (_Float16*)(ws + 262144);   // 128 KB
  float*    paths  = (float*)   (ws + 393216);   // 256 KB
  _Float16* Rt     = (_Float16*)(ws + 655360);   // 2 MB
  unsigned* maskR  = (unsigned*)(ws + 2752512);  // 128 KB
  unsigned* maskRT = (unsigned*)(ws + 2883584);  // 128 KB
  int*      idx    = (int*)     (ws + 3014656);  // 256 B
  float*    part   = (float*)   (ws + 3014912);  // 1 KB

  float* out = (float*)d_out;               // [0]=loss, [1..]=probs (64*1024)

  k_sigmoid   <<<256, 256, 0, stream>>>(h, x, xh, BB * CC);
  k_convR     <<<4096, 256, 0, stream>>>(R, Rt, maskR);
  k_maskT     <<<4096, 256, 0, stream>>>(Rt, maskRT);
  k_paths_wmma<<<64, 128, 0, stream>>>(xh, Rt, paths);
  k_argmax    <<<64, 256, 0, stream>>>(paths, idx);
  dim3 g(4, BB);
  k_loss      <<<g, 256, 0, stream>>>(x, y, maskR, maskRT, idx, out, part);
  k_final     <<<1, 256, 0, stream>>>(part, out);
}
